// GNN_4896262717835
// MI455X (gfx1250) — matrix operand order repaired and checked by reading, not compiled, not run on hardware
//
#include <hip/hip_runtime.h>

// ---------------------------------------------------------------------------
// GNN forward for MI455X (gfx1250, wave32).
// All 128-wide GEMMs run on v_wmma_f32_16x16x32_bf16: one wave = 16 rows x 128
// cols (8 x v8f accumulators). Activations kept in bf16 (h fits in 192MB L2,
// so edge gathers are L2 traffic); accumulation in f32. Edge MLP chains are
// fully fused per 16-edge tile; scatter-mean via global f32 atomics.
// This round: native (__bf16) casts instead of manual RNE bit-twiddling, and
// coalesced b128 stores of edge states via the LDS tile.
// ---------------------------------------------------------------------------

#define LDK 136   // padded K stride (ushorts) for per-wave LDS transpose tile

typedef __attribute__((ext_vector_type(8)))  unsigned short v8u;
typedef __attribute__((ext_vector_type(16))) unsigned short v16u_t;
typedef __attribute__((ext_vector_type(16))) __bf16        v16bf;
typedef __attribute__((ext_vector_type(8)))  float         v8f;

union FragU { v16u_t u; v16bf b; };

__device__ __forceinline__ unsigned short f2bf(float f) {
  __bf16 b = (__bf16)f;                      // native v_cvt_*bf16_f32 on gfx1250
  return __builtin_bit_cast(unsigned short, b);
}
__device__ __forceinline__ float silu_f(float x) { return x / (1.0f + __expf(-x)); }

__device__ __forceinline__ void zero_acc(v8f acc[8]) {
#pragma unroll
  for (int nt = 0; nt < 8; ++nt) {
    v8f z = {0.f,0.f,0.f,0.f,0.f,0.f,0.f,0.f};
    acc[nt] = z;
  }
}

__device__ __forceinline__ v16bf mk_frag(v8u lo, v8u hi) {
  FragU f;
#pragma unroll
  for (int i = 0; i < 8; ++i) { f.u[i] = lo[i]; f.u[i + 8] = hi[i]; }
  return f.b;
}

// A fragment (16x32, bf16): lane holds row m=lane%16; elements 0..7 -> K=half*8+e,
// elements 8..15 -> K=16+half*8+(e-8). Two 16B loads from a K-contiguous row.
__device__ __forceinline__ v16bf frag_row(const unsigned short* row, int kt, int half) {
  v8u lo = *(const v8u*)(row + kt + half * 8);
  v8u hi = *(const v8u*)(row + kt + 16 + half * 8);
  return mk_frag(lo, hi);
}

// One K=32 WMMA step against all 8 N-tiles (128 output columns).
// B fragment (32x16, bf16): lane holds col n=lane%16, elements 0..7 -> K=half*8+e, 8..15 -> K=16+half*8+(e-8)
// in the pre-transposed Wt[N][Kpad] bf16 weight image.
__device__ __forceinline__ void wmma_step(v8f acc[8], v16bf a,
                                          const unsigned short* __restrict__ Wt,
                                          int ldw, int koff, int laneN, int half) {
#pragma unroll
  for (int nt = 0; nt < 8; ++nt) {
    const unsigned short* wr = Wt + (size_t)(nt * 16 + laneN) * ldw + koff + half * 8;
    v8u lo = *(const v8u*)wr;
    v8u hi = *(const v8u*)(wr + 16);
    acc[nt] = __builtin_amdgcn_wmma_f32_16x16x32_bf16(
        false, a, false, mk_frag(lo, hi), (short)0, acc[nt], false, false);
  }
}

__device__ __forceinline__ void gemm128(v8f acc[8], const unsigned short* arow,
                                        const unsigned short* Wt, int ldw, int koff,
                                        int laneN, int half) {
#pragma unroll
  for (int kt = 0; kt < 128; kt += 32)
    wmma_step(acc, frag_row(arow, kt, half), Wt, ldw, koff + kt, laneN, half);
}

__device__ __forceinline__ void bias_silu_acc(v8f acc[8], const float* __restrict__ b,
                                              int laneN) {
#pragma unroll
  for (int nt = 0; nt < 8; ++nt) {
    float bb = b[nt * 16 + laneN];
#pragma unroll
    for (int v = 0; v < 8; ++v) acc[nt][v] = silu_f(acc[nt][v] + bb);
  }
}

// D(16x16 f32) layout: lane holds col n=lane%16, rows v+8*half. Store (bias+act)
// result as bf16 row-major into the per-wave LDS tile for the next GEMM's A.
__device__ __forceinline__ void store_tile(unsigned short* tile, v8f acc[8],
                                           const float* b, bool act,
                                           int laneN, int half) {
#pragma unroll
  for (int nt = 0; nt < 8; ++nt) {
    float bb = b ? b[nt * 16 + laneN] : 0.0f;
#pragma unroll
    for (int v = 0; v < 8; ++v) {
      float xv = acc[nt][v] + bb;
      if (act) xv = silu_f(xv);
      tile[(v + 8 * half) * LDK + nt * 16 + laneN] = f2bf(xv);
    }
  }
}

// ---------------------------------------------------------------------------
// prep: W[K][N] f32 -> Wt[N][Kpad] bf16 (zero pad rows k >= K)
// ---------------------------------------------------------------------------
__global__ __launch_bounds__(256) void prep_wt_kernel(const float* __restrict__ W,
                                                      unsigned short* __restrict__ Wt,
                                                      int K, int Nn, int Kpad) {
  int i = blockIdx.x * 256 + threadIdx.x;
  if (i >= Nn * Kpad) return;
  int n = i / Kpad, k = i - n * Kpad;
  Wt[i] = (k < K) ? f2bf(W[(size_t)k * Nn + n]) : (unsigned short)0;
}

__global__ __launch_bounds__(256) void fillz_kernel(float* __restrict__ p, long n) {
  long i = (long)blockIdx.x * 256 + threadIdx.x;
  if (i < n) p[i] = 0.0f;
}

__global__ __launch_bounds__(256) void count_kernel(const int* __restrict__ eidx,
                                                    float* __restrict__ cnt, int E_) {
  int e = blockIdx.x * 256 + threadIdx.x;
  if (e < E_) atomicAdd(&cnt[eidx[E_ + e]], 1.0f);
}

// ---------------------------------------------------------------------------
// f_edge: gaussian-expand (computed in-register per A fragment) -> 256->128
// -> 128->128 -> 128->128, SiLU each; coalesced b128 stores of bf16 edge states
// through the LDS tile.
// ---------------------------------------------------------------------------
__global__ __launch_bounds__(256) void fedge_kernel(
    const float* __restrict__ exn,
    const unsigned short* __restrict__ W0, const float* __restrict__ b0,
    const unsigned short* __restrict__ W1, const float* __restrict__ b1,
    const unsigned short* __restrict__ W2, const float* __restrict__ b2,
    unsigned short* __restrict__ est, int E_) {
  __shared__ unsigned short act[8][16 * LDK];
  int wave = threadIdx.x >> 5, lane = threadIdx.x & 31;
  int half = lane >> 4, laneN = lane & 15;
  long eBase = (long)blockIdx.x * 128 + wave * 16;
  if (eBase >= E_) return;
  long eA = eBase + laneN; if (eA > E_ - 1) eA = E_ - 1;
  float d0 = exn[eA * 4 + 0], d1 = exn[eA * 4 + 1];
  float d2 = exn[eA * 4 + 2], d3 = exn[eA * 4 + 3];
  unsigned short* tile = &act[wave][0];

  v8f acc[8]; zero_acc(acc);
#pragma unroll
  for (int kt = 0; kt < 256; kt += 32) {   // K = 4*G = 256
    FragU f;
#pragma unroll
    for (int i = 0; i < 16; ++i) {
      int k = kt + ((i < 8) ? (half * 8 + i) : (16 + half * 8 + (i - 8)));
      float dv = (k < 64) ? d0 : (k < 128) ? d1 : (k < 192) ? d2 : d3;
      float mu = (float)(k & 63) * (1.0f / 63.0f);
      float t = dv - mu;
      f.u[i] = f2bf(__expf(-t * t * 1984.5f));   // 1/(2*sigma^2), sigma = 1/63
    }
    wmma_step(acc, f.b, W0, 256, kt, laneN, half);
  }
  store_tile(tile, acc, b0, true, laneN, half);
  zero_acc(acc); gemm128(acc, tile + laneN * LDK, W1, 128, 0, laneN, half);
  store_tile(tile, acc, b1, true, laneN, half);
  zero_acc(acc); gemm128(acc, tile + laneN * LDK, W2, 128, 0, laneN, half);
  store_tile(tile, acc, b2, true, laneN, half);

  // coalesced copy tile -> est (16 rows x 128 bf16, 16B chunks)
#pragma unroll
  for (int i = 0; i < 8; ++i) {
    int j = lane + 32 * i;                 // 0..255 chunk id
    int row = j >> 4, cc = j & 15;
    long e = eBase + row;
    if (e < E_) {
      v8u vdat = *(const v8u*)(tile + row * LDK + cc * 8);
      *(v8u*)(est + e * 128 + cc * 8) = vdat;
    }
  }
}

// ---------------------------------------------------------------------------
// f_node: x[N,16] (K padded to 32) -> 128 -> 128, SiLU each; write h f32 + bf16
// ---------------------------------------------------------------------------
__global__ __launch_bounds__(256) void fnode_kernel(
    const float* __restrict__ x,
    const unsigned short* __restrict__ W0, const float* __restrict__ b0,
    const unsigned short* __restrict__ W1, const float* __restrict__ b1,
    float* __restrict__ h, unsigned short* __restrict__ hbf, int N_) {
  __shared__ unsigned short act[8][16 * LDK];
  int wave = threadIdx.x >> 5, lane = threadIdx.x & 31;
  int half = lane >> 4, laneN = lane & 15;
  long nBase = (long)blockIdx.x * 128 + wave * 16;
  if (nBase >= N_) return;
  long nA = nBase + laneN; if (nA > N_ - 1) nA = N_ - 1;
  const float* xr = x + nA * 16;
  FragU f;
#pragma unroll
  for (int i = 0; i < 8; ++i) { f.u[i] = f2bf(xr[half * 8 + i]); f.u[8 + i] = 0; }
  unsigned short* tile = &act[wave][0];
  v8f acc[8]; zero_acc(acc);
  wmma_step(acc, f.b, W0, 32, 0, laneN, half);     // single K=32 step (padded)
  store_tile(tile, acc, b0, true, laneN, half);
  zero_acc(acc); gemm128(acc, tile + laneN * LDK, W1, 128, 0, laneN, half);
#pragma unroll
  for (int nt = 0; nt < 8; ++nt) {
    float bb = b1[nt * 16 + laneN];
#pragma unroll
    for (int v = 0; v < 8; ++v) {
      long r = nBase + v + 8 * half;
      if (r < N_) {
        float y = silu_f(acc[nt][v] + bb);
        h[r * 128 + nt * 16 + laneN] = y;
        hbf[r * 128 + nt * 16 + laneN] = f2bf(y);
      }
    }
  }
}

// ---------------------------------------------------------------------------
// Per-layer edge kernel, fully fused per 16-edge tile:
//   mes  = silu(silu(h[src]@Wm0+bm0)@Wm1+bm1)
//   int  = silu(silu([h[src]|est|h[dst]]@Wi0+bi0)@Wi1+bi1) * cos(pi/2*nd)
//   atomically scatter (int*mes) into sums[dst]
// A fragments gather straight from global bf16 rows (L2-resident).
// ---------------------------------------------------------------------------
__global__ __launch_bounds__(256) void layer_edge_kernel(
    const int* __restrict__ eidx, const float* __restrict__ exn,
    const unsigned short* __restrict__ hbf, const unsigned short* __restrict__ est,
    const unsigned short* __restrict__ WtI0, const float* __restrict__ bI0,
    const unsigned short* __restrict__ WtI1, const float* __restrict__ bI1,
    const unsigned short* __restrict__ WtM0, const float* __restrict__ bM0,
    const unsigned short* __restrict__ WtM1, const float* __restrict__ bM1,
    float* __restrict__ sums, int E_) {
  __shared__ unsigned short act[8][16 * LDK];
  int wave = threadIdx.x >> 5, lane = threadIdx.x & 31;
  int half = lane >> 4, laneN = lane & 15;
  long eBase = (long)blockIdx.x * 128 + wave * 16;
  if (eBase >= E_) return;
  long eA = eBase + laneN; if (eA > E_ - 1) eA = E_ - 1;
  int sA = eidx[eA];
  int dA = eidx[E_ + eA];
  const unsigned short* hsR = hbf + (size_t)sA * 128;
  const unsigned short* hdR = hbf + (size_t)dA * 128;
  const unsigned short* esR = est + (size_t)eA * 128;
  __builtin_prefetch(esR, 0, 0);               // global_prefetch_b8
  unsigned short* tile = &act[wave][0];

  // ---- message path: f_mes(h[src]) ----
  v8f accM[8]; zero_acc(accM);
  gemm128(accM, hsR, WtM0, 128, 0, laneN, half);
  store_tile(tile, accM, bM0, true, laneN, half);
  zero_acc(accM); gemm128(accM, tile + laneN * LDK, WtM1, 128, 0, laneN, half);
  bias_silu_acc(accM, bM1, laneN);             // final mes values stay in regs

  // ---- interaction path: f_int(concat) over K=384 as 3 K=128 segments ----
  v8f accI[8]; zero_acc(accI);
  gemm128(accI, hsR, WtI0, 384, 0,   laneN, half);
  gemm128(accI, esR, WtI0, 384, 128, laneN, half);
  gemm128(accI, hdR, WtI0, 384, 256, laneN, half);
  store_tile(tile, accI, bI0, true, laneN, half);
  zero_acc(accI); gemm128(accI, tile + laneN * LDK, WtI1, 128, 0, laneN, half);
  bias_silu_acc(accI, bI1, laneN);

  // ---- decay, elementwise product, scatter-add into sums[dst] ----
#pragma unroll
  for (int v = 0; v < 8; ++v) {
    long e = eBase + v + 8 * half;
    if (e < E_) {
      float dcy = __cosf(1.5707963267948966f * exn[e * 4 + 3]);
      int dnode = eidx[E_ + e];
      float* srow = sums + (size_t)dnode * 128;
#pragma unroll
      for (int nt = 0; nt < 8; ++nt)
        atomicAdd(&srow[nt * 16 + laneN], accI[nt][v] * dcy * accM[nt][v]);
    }
  }
}

// ---------------------------------------------------------------------------
// Per-layer node kernel: m_t = mes(h)@ind_mask; u = m_t + sums/max(cnt,1);
// h2 = silu(u@Wu0+bu0)@Wu1+bu1; write h2 + accumulate BN column stats.
// ---------------------------------------------------------------------------
__global__ __launch_bounds__(256) void node_update_kernel(
    const unsigned short* __restrict__ hbf, const float* __restrict__ sums,
    const float* __restrict__ cnt,
    const unsigned short* __restrict__ WtM0, const float* __restrict__ bM0,
    const unsigned short* __restrict__ WtM1, const float* __restrict__ bM1,
    const unsigned short* __restrict__ WtIND,
    const unsigned short* __restrict__ WtU0, const float* __restrict__ bU0,
    const unsigned short* __restrict__ WtU1, const float* __restrict__ bU1,
    float* __restrict__ h2, float* __restrict__ colSum, float* __restrict__ colSq,
    int N_) {
  __shared__ unsigned short act[8][16 * LDK];
  int wave = threadIdx.x >> 5, lane = threadIdx.x & 31;
  int half = lane >> 4, laneN = lane & 15;
  long nBase = (long)blockIdx.x * 128 + wave * 16;
  if (nBase >= N_) return;
  long nA = nBase + laneN; if (nA > N_ - 1) nA = N_ - 1;
  const unsigned short* hr = hbf + (size_t)nA * 128;
  unsigned short* tile = &act[wave][0];

  v8f acc[8]; zero_acc(acc);
  gemm128(acc, hr, WtM0, 128, 0, laneN, half);
  store_tile(tile, acc, bM0, true, laneN, half);
  zero_acc(acc); gemm128(acc, tile + laneN * LDK, WtM1, 128, 0, laneN, half);
  store_tile(tile, acc, bM1, true, laneN, half);
  zero_acc(acc); gemm128(acc, tile + laneN * LDK, WtIND, 128, 0, laneN, half); // m_t

  // + incoming (scatter-mean)
#pragma unroll
  for (int nt = 0; nt < 8; ++nt) {
    int col = nt * 16 + laneN;
#pragma unroll
    for (int v = 0; v < 8; ++v) {
      long r = nBase + v + 8 * half;
      if (r < N_) {
        float c = fmaxf(cnt[r], 1.0f);
        acc[nt][v] += sums[(size_t)r * 128 + col] / c;
      }
    }
  }
  store_tile(tile, acc, nullptr, false, laneN, half);        // u (raw, bf16)
  zero_acc(acc); gemm128(acc, tile + laneN * LDK, WtU0, 128, 0, laneN, half);
  store_tile(tile, acc, bU0, true, laneN, half);
  zero_acc(acc); gemm128(acc, tile + laneN * LDK, WtU1, 128, 0, laneN, half);

#pragma unroll
  for (int nt = 0; nt < 8; ++nt) {
    int col = nt * 16 + laneN;
    float bb = bU1[col], s = 0.f, q = 0.f;
#pragma unroll
    for (int v = 0; v < 8; ++v) {
      long r = nBase + v + 8 * half;
      if (r < N_) {
        float xv = acc[nt][v] + bb;
        h2[(size_t)r * 128 + col] = xv;
        s += xv; q += xv * xv;
      }
    }
    atomicAdd(&colSum[col], s);
    atomicAdd(&colSq[col], q);
  }
}

__global__ void bn_finalize_kernel(const float* __restrict__ colSum,
                                   const float* __restrict__ colSq,
                                   const float* __restrict__ gamma,
                                   const float* __restrict__ beta,
                                   float* __restrict__ scaleB,
                                   float* __restrict__ shiftB, int N_) {
  int c = threadIdx.x;
  if (c < 128) {
    float mu = colSum[c] / (float)N_;
    float var = colSq[c] / (float)N_ - mu * mu;
    float sc = gamma[c] * rsqrtf(var + 1e-5f);
    scaleB[c] = sc;
    shiftB[c] = beta[c] - mu * sc;
  }
}

__global__ __launch_bounds__(256) void bn_apply_kernel(
    const float* __restrict__ h2, const float* __restrict__ scaleB,
    const float* __restrict__ shiftB, float* __restrict__ h,
    unsigned short* __restrict__ hbf, long total) {
  long i = (long)blockIdx.x * 256 + threadIdx.x;
  if (i < total) {
    int c = (int)(i & 127);
    float y = silu_f(h2[i] * scaleB[c] + shiftB[c]) + h[i];   // + residual
    h[i] = y;
    hbf[i] = f2bf(y);
  }
}

// ---------------------------------------------------------------------------
// target head 128->64->32->1 (tiny; VALU with LDS-cached weights)
// ---------------------------------------------------------------------------
__global__ __launch_bounds__(256) void target_kernel(
    const float* __restrict__ h,
    const float* __restrict__ W0, const float* __restrict__ b0,
    const float* __restrict__ W1, const float* __restrict__ b1,
    const float* __restrict__ W2, const float* __restrict__ b2,
    float* __restrict__ out, int N_) {
  __shared__ float sW0[128 * 64];
  __shared__ float sW1[64 * 32];
  __shared__ float sW2[32];
  __shared__ float sb0[64], sb1[32];
  for (int i = threadIdx.x; i < 128 * 64; i += 256) sW0[i] = W0[i];
  for (int i = threadIdx.x; i < 64 * 32; i += 256) sW1[i] = W1[i];
  if (threadIdx.x < 32) sW2[threadIdx.x] = W2[threadIdx.x];
  if (threadIdx.x < 64) sb0[threadIdx.x] = b0[threadIdx.x];
  if (threadIdx.x < 32) sb1[threadIdx.x] = b1[threadIdx.x];
  __syncthreads();
  long n = (long)blockIdx.x * 256 + threadIdx.x;
  if (n >= N_) return;
  float t0[64];
#pragma unroll
  for (int j = 0; j < 64; ++j) t0[j] = sb0[j];
  for (int k = 0; k < 128; ++k) {
    float hv = h[n * 128 + k];
#pragma unroll
    for (int j = 0; j < 64; ++j) t0[j] += hv * sW0[k * 64 + j];
  }
#pragma unroll
  for (int j = 0; j < 64; ++j) t0[j] = silu_f(t0[j]);
  float t1[32];
#pragma unroll
  for (int j = 0; j < 32; ++j) t1[j] = sb1[j];
  for (int k = 0; k < 64; ++k) {
    float v = t0[k];
#pragma unroll
    for (int j = 0; j < 32; ++j) t1[j] += v * sW1[k * 32 + j];
  }
  float y = b2[0];
#pragma unroll
  for (int k = 0; k < 32; ++k) y += silu_f(t1[k]) * sW2[k];
  out[n] = y;
}

// ---------------------------------------------------------------------------
// Host-side orchestration.
// Input flattening (jax tree order, dict keys sorted):
//  0 edge_index, 1 ex_norm_displacement,
//  2..7 f_edge{W,b}x3, 8..11 f_node{W,b}x2, 12..17 f_target{W,b}x3,
//  per layer l (base=18+15l): +0 bn.beta,+1 bn.gamma, +2..5 f_int, +6..9 f_mes,
//  +10..13 f_upd, +14 ind_imp_mask;  63 x
// ---------------------------------------------------------------------------
extern "C" void kernel_launch(void* const* d_in, const int* in_sizes, int n_in,
                              void* d_out, int out_size, void* d_ws, size_t ws_size,
                              hipStream_t stream) {
  (void)n_in; (void)out_size; (void)ws_size;
  const int E = in_sizes[0] / 2;
  const int N = in_sizes[63] / 16;
  const int*   eidx = (const int*)d_in[0];
  const float* exn  = (const float*)d_in[1];
  const float* x    = (const float*)d_in[63];

  char* wsb = (char*)d_ws;
  size_t off = 0;
  auto carve = [&](size_t bytes) -> char* {
    char* p = wsb + off;
    off = (off + bytes + 255) & ~(size_t)255;
    return p;
  };
  unsigned short* est  = (unsigned short*)carve((size_t)E * 128 * 2);
  float*          h    = (float*)carve((size_t)N * 128 * 4);
  unsigned short* hbf  = (unsigned short*)carve((size_t)N * 128 * 2);
  float*          h2   = (float*)carve((size_t)N * 128 * 4);
  float*          sums = (float*)carve((size_t)N * 128 * 4);
  float*          cnt  = (float*)carve((size_t)N * 4);
  float*          colstats = (float*)carve(256 * 4);
  float*          scaleB   = (float*)carve(128 * 4);
  float*          shiftB   = (float*)carve(128 * 4);
  float* colSum = colstats;
  float* colSq  = colstats + 128;

  auto prep = [&](int idx, int K, int Kpad) -> unsigned short* {
    unsigned short* wt = (unsigned short*)carve((size_t)128 * Kpad * 2);
    int tot = 128 * Kpad;
    prep_wt_kernel<<<(tot + 255) / 256, 256, 0, stream>>>(
        (const float*)d_in[idx], wt, K, 128, Kpad);
    return wt;
  };

  unsigned short* WtE0 = prep(2, 256, 256);
  unsigned short* WtE1 = prep(4, 128, 128);
  unsigned short* WtE2 = prep(6, 128, 128);
  unsigned short* WtN0 = prep(8, 16, 32);      // K=16 zero-padded to 32
  unsigned short* WtN1 = prep(10, 128, 128);
  unsigned short* WtI0[3]; unsigned short* WtI1[3];
  unsigned short* WtM0[3]; unsigned short* WtM1[3];
  unsigned short* WtU0[3]; unsigned short* WtU1[3];
  unsigned short* WtIN[3];
  for (int l = 0; l < 3; ++l) {
    int base = 18 + 15 * l;
    WtI0[l] = prep(base + 2, 384, 384);
    WtI1[l] = prep(base + 4, 128, 128);
    WtM0[l] = prep(base + 6, 128, 128);
    WtM1[l] = prep(base + 8, 128, 128);
    WtU0[l] = prep(base + 10, 128, 128);
    WtU1[l] = prep(base + 12, 128, 128);
    WtIN[l] = prep(base + 14, 128, 128);
  }

  // degree counts (dst is layer-invariant): once
  fillz_kernel<<<(N + 255) / 256, 256, 0, stream>>>(cnt, (long)N);
  count_kernel<<<(E + 255) / 256, 256, 0, stream>>>(eidx, cnt, E);

  // initial edge states + node embedding
  fedge_kernel<<<(E + 127) / 128, 256, 0, stream>>>(
      exn, WtE0, (const float*)d_in[3], WtE1, (const float*)d_in[5],
      WtE2, (const float*)d_in[7], est, E);
  fnode_kernel<<<(N + 127) / 128, 256, 0, stream>>>(
      x, WtN0, (const float*)d_in[9], WtN1, (const float*)d_in[11], h, hbf, N);

  for (int l = 0; l < 3; ++l) {
    int base = 18 + 15 * l;
    const float* bnBeta  = (const float*)d_in[base + 0];
    const float* bnGamma = (const float*)d_in[base + 1];
    fillz_kernel<<<(int)(((long)N * 128 + 255) / 256), 256, 0, stream>>>(
        sums, (long)N * 128);
    fillz_kernel<<<1, 256, 0, stream>>>(colstats, 256L);

    layer_edge_kernel<<<(E + 127) / 128, 256, 0, stream>>>(
        eidx, exn, hbf, est,
        WtI0[l], (const float*)d_in[base + 3], WtI1[l], (const float*)d_in[base + 5],
        WtM0[l], (const float*)d_in[base + 7], WtM1[l], (const float*)d_in[base + 9],
        sums, E);

    node_update_kernel<<<(N + 127) / 128, 256, 0, stream>>>(
        hbf, sums, cnt,
        WtM0[l], (const float*)d_in[base + 7], WtM1[l], (const float*)d_in[base + 9],
        WtIN[l],
        WtU0[l], (const float*)d_in[base + 11], WtU1[l], (const float*)d_in[base + 13],
        h2, colSum, colSq, N);

    bn_finalize_kernel<<<1, 128, 0, stream>>>(colSum, colSq, bnGamma, bnBeta,
                                              scaleB, shiftB, N);
    bn_apply_kernel<<<(int)(((long)N * 128 + 255) / 256), 256, 0, stream>>>(
        h2, scaleB, shiftB, h, hbf, (long)N * 128);
  }

  target_kernel<<<(N + 255) / 256, 256, 0, stream>>>(
      h, (const float*)d_in[12], (const float*)d_in[13],
      (const float*)d_in[14], (const float*)d_in[15],
      (const float*)d_in[16], (const float*)d_in[17], (float*)d_out, N);
}
